// DecoderTransformer_1073741824003
// MI455X (gfx1250) — matrix operand order repaired and checked by reading, not compiled, not run on hardware
//
#include <hip/hip_runtime.h>
#include <hip/hip_bf16.h>

#define E_  1024
#define NH_ 16
#define HD_ 64
#define FF_ 4096
#define L_  6
#define V_  32000
#define B_  32
#define T_  100
#define S_  99
#define M_  (B_*S_)   /* 3168, multiple of 16 */
#define MP_ 3200      /* padded to 25*128 */

typedef unsigned short u16;
typedef unsigned long long u64;
typedef __attribute__((ext_vector_type(16))) __bf16 v16bf;
typedef __attribute__((ext_vector_type(8)))  float  v8f;

__device__ __forceinline__ u16 f2bf(float f) {
  unsigned u = __float_as_uint(f);
  return (u16)((u + 0x7FFFu + ((u >> 16) & 1u)) >> 16);
}

// ---------------------------------------------------------------- conversions
__global__ void cvt_f32_bf16(const float* __restrict__ in, u16* __restrict__ out, long n) {
  long i = (long)blockIdx.x * 1024 + (long)threadIdx.x * 4;
#pragma unroll
  for (int j = 0; j < 4; ++j) {
    long k = i + j;
    if (k < n) out[k] = f2bf(in[k]);
  }
}

// ------------------------------------------------------- embedding + pos enc
__global__ void embed_kernel(const int* __restrict__ captions,
                             const float* __restrict__ tab,
                             const float* __restrict__ pos,
                             float* __restrict__ x, u16* __restrict__ xh) {
  int m = blockIdx.x;          // 0..MP_-1
  int c0 = threadIdx.x * 4;
  if (m < M_) {
    int b = m / S_, s = m % S_;
    int tok = captions[b * T_ + s];
    const float* er = tab + (size_t)tok * E_;
    const float* pr = pos + (size_t)s * E_;
#pragma unroll
    for (int j = 0; j < 4; ++j) {
      int c = c0 + j;
      float v = er[c] + pr[c];
      size_t o = (size_t)m * E_ + c;
      x[o] = v; xh[o] = f2bf(v);
    }
  } else {
#pragma unroll
    for (int j = 0; j < 4; ++j) {
      size_t o = (size_t)m * E_ + c0 + j;
      x[o] = 0.f; xh[o] = 0;
    }
  }
}

// pad features (B_ x E_) into 128 x E_ bf16 (zeros beyond B_)
__global__ void feat_pad_kernel(const float* __restrict__ f, u16* __restrict__ fh) {
  int m = blockIdx.x;          // 0..127
  int c0 = threadIdx.x * 4;
#pragma unroll
  for (int j = 0; j < 4; ++j) {
    int c = c0 + j;
    size_t o = (size_t)m * E_ + c;
    fh[o] = (m < B_) ? f2bf(f[o]) : (u16)0;
  }
}

__global__ void zero_t0_kernel(float* __restrict__ out) {
  long i = (long)blockIdx.x * 256 + threadIdx.x;
  if (i < (long)B_ * V_) {
    long b = i / V_, v = i % V_;
    out[b * (long)T_ * V_ + v] = 0.f;
  }
}

// ------------------------------------------------ async global->LDS staging
// Stage one 128x32 bf16 A tile and one 128x32 bf16 W tile with
// GLOBAL_LOAD_ASYNC_TO_LDS_B128 (GV mode, tracked by ASYNCcnt).
// Each of 256 threads issues 2 chunks for A and 2 for B -> 4 async
// instructions *per wave* per tile (wave32: one instr moves 32x16B).
__device__ __forceinline__ void stage_tile_async(
    const u16* __restrict__ gA, const u16* __restrict__ gW,
    u16* sA, u16* sB, int tid, int K, int k0, int bm128, int bn128) {
#pragma unroll
  for (int c = 0; c < 2; ++c) {
    int id = tid + c * 256;          // 0..511
    int row = id >> 2;               // 0..127
    int kc = (id & 3) << 3;          // 0,8,16,24 halves
    unsigned la = (unsigned)(size_t)(sA + row * 40 + kc);
    u64 ga = (u64)(size_t)(gA + (size_t)(bm128 + row) * K + k0 + kc);
    asm volatile("global_load_async_to_lds_b128 %0, %1, off"
                 :: "v"(la), "v"(ga) : "memory");
    unsigned lb = (unsigned)(size_t)(sB + row * 40 + kc);
    u64 gb = (u64)(size_t)(gW + (size_t)(bn128 + row) * K + k0 + kc);
    asm volatile("global_load_async_to_lds_b128 %0, %1, off"
                 :: "v"(lb), "v"(gb) : "memory");
  }
}

// -------------------------------------------------------------- WMMA GEMM
// C(Mpad x N) = A(Mpad x K, bf16) * W^T (W is N x K row-major, bf16) + bias
// MODE 0: f32 out   MODE 1: ReLU -> bf16 out   MODE 2: scatter logits out[:,1:,:]
template <int MODE>
__global__ __launch_bounds__(256) void gemm_bf16_wmma(
    const u16* __restrict__ A, const u16* __restrict__ Wt,
    const float* __restrict__ bias, float* __restrict__ C,
    u16* __restrict__ C16, int N, int K, int Mreal) {
  __shared__ __align__(16) u16 As[2][128 * 40];  // stride 40 halves = 80B
  __shared__ __align__(16) u16 Bs[2][128 * 40];

  const int tid = threadIdx.x;
  const int lane = tid & 31;
  const int wid = tid >> 5;
  const int wm = (wid & 3) * 32;   // 4 waves along M -> 128
  const int wn = (wid >> 2) * 64;  // 2 waves along N -> 128
  const int bm = blockIdx.y, bn = blockIdx.x;
  const int lr = lane & 15;

  v8f zero = {0.f, 0.f, 0.f, 0.f, 0.f, 0.f, 0.f, 0.f};
  v8f acc[2][4];
#pragma unroll
  for (int mi = 0; mi < 2; ++mi)
#pragma unroll
    for (int ni = 0; ni < 4; ++ni) acc[mi][ni] = zero;

  const int nk = K >> 5;
  stage_tile_async(A, Wt, As[0], Bs[0], tid, K, 0, bm * 128, bn * 128);

  for (int kt = 0; kt < nk; ++kt) {
    const int cur = kt & 1;
    if (kt + 1 < nk) {
      // prefetch next tile into the other buffer, then wait only for the
      // 4 older per-wave async ops (this tile); loads complete in order.
      stage_tile_async(A, Wt, As[cur ^ 1], Bs[cur ^ 1], tid, K,
                       (kt + 1) << 5, bm * 128, bn * 128);
      asm volatile("s_wait_asynccnt 0x4" ::: "memory");
    } else {
      asm volatile("s_wait_asynccnt 0x0" ::: "memory");
    }
    __syncthreads();  // all waves' async writes for tile kt visible

    // A fragments: lanes 0-15 hold K {0..7,16..23}, lanes 16-31 {8..15,24..31}
    v16bf afr[2];
#pragma unroll
    for (int mi = 0; mi < 2; ++mi) {
      const u16* rb = &As[cur][(wm + mi * 16 + lr) * 40];
      int klo = (lane & 16) ? 8 : 0;
      union { v16bf v; uint4 q[2]; } u;
      u.q[0] = *reinterpret_cast<const uint4*>(rb + klo);
      u.q[1] = *reinterpret_cast<const uint4*>(rb + klo + 16);
      afr[mi] = u.v;
    }
    // B fragments: lanes 0-15 hold K {0..7,16..23}, lanes 16-31 {8..15,24..31}
    v16bf bfr[4];
#pragma unroll
    for (int ni = 0; ni < 4; ++ni) {
      const u16* rb = &Bs[cur][(wn + ni * 16 + lr) * 40];
      int klo = (lane & 16) ? 8 : 0;
      union { v16bf v; uint4 q[2]; } u;
      u.q[0] = *reinterpret_cast<const uint4*>(rb + klo);
      u.q[1] = *reinterpret_cast<const uint4*>(rb + klo + 16);
      bfr[ni] = u.v;
    }

#pragma unroll
    for (int mi = 0; mi < 2; ++mi)
#pragma unroll
      for (int ni = 0; ni < 4; ++ni)
        acc[mi][ni] = __builtin_amdgcn_wmma_f32_16x16x32_bf16(
            false, afr[mi], false, bfr[ni], (short)0, acc[mi][ni], false, false);
    __syncthreads();  // done reading buffer cur; next iter may overwrite it
  }

  // epilogue: C layout — VGPR r: lanes 0-15 => M=r, lanes 16-31 => M=r+8
  const int rowadd = (lane & 16) ? 8 : 0;
#pragma unroll
  for (int ni = 0; ni < 4; ++ni) {
    int n = bn * 128 + wn + ni * 16 + lr;
    float bv = bias[n];
#pragma unroll
    for (int mi = 0; mi < 2; ++mi) {
#pragma unroll
      for (int r = 0; r < 8; ++r) {
        int m = bm * 128 + wm + mi * 16 + r + rowadd;
        float v = acc[mi][ni][r] + bv;
        if constexpr (MODE == 0) {
          C[(size_t)m * N + n] = v;
        } else if constexpr (MODE == 1) {
          v = v > 0.f ? v : 0.f;
          C16[(size_t)m * N + n] = f2bf(v);
        } else {  // MODE == 2: logits scatter into out[:, 1:, :]
          if (m < Mreal) {
            int b = m / S_, s = m % S_;
            C[((size_t)b * T_ + s + 1) * N + n] = v;
          }
        }
      }
    }
  }
}

// ---------------------------------------------- fused causal attention (f32)
// qkv: (M_ x 3E) rows [q|k|v]; one block per (b,h,q); softmax over k<=q
__global__ void attn_kernel(const float* __restrict__ qkv,
                            float* __restrict__ ctx, u16* __restrict__ ctxh) {
  const int q = blockIdx.x, h = blockIdx.y, b = blockIdx.z;
  const int tid = threadIdx.x;
  __shared__ float qs[HD_];
  __shared__ float sc[128];
  __shared__ float inv_sum;

  const size_t rq = (size_t)(b * S_ + q) * (3 * E_);
  if (tid < HD_) qs[tid] = qkv[rq + h * HD_ + tid];
  __syncthreads();

  const int nk = q + 1;
  for (int k = tid; k < nk; k += 128) {
    const float* kr = qkv + (size_t)(b * S_ + k) * (3 * E_) + E_ + h * HD_;
    float d = 0.f;
#pragma unroll 8
    for (int j = 0; j < HD_; ++j) d += qs[j] * kr[j];
    sc[k] = d * 0.125f;  // 1/sqrt(64)
  }
  __syncthreads();
  if (tid == 0) {
    float mx = -1e30f;
    for (int k = 0; k < nk; ++k) mx = sc[k] > mx ? sc[k] : mx;
    float sm = 0.f;
    for (int k = 0; k < nk; ++k) { float e = __expf(sc[k] - mx); sc[k] = e; sm += e; }
    inv_sum = 1.f / sm;
  }
  __syncthreads();
  if (tid < HD_) {
    float a = 0.f;
    for (int k = 0; k < nk; ++k)
      a += sc[k] * qkv[(size_t)(b * S_ + k) * (3 * E_) + 2 * E_ + h * HD_ + tid];
    a *= inv_sum;
    size_t o = (size_t)(b * S_ + q) * E_ + h * HD_ + tid;
    ctx[o] = a; ctxh[o] = f2bf(a);
  }
}

// ------------------------------------------------- residual add + layernorm
__global__ void add_ln_kernel(const float* __restrict__ xres,
                              const float* __restrict__ y, int ybcast,
                              const float* __restrict__ g, const float* __restrict__ bb,
                              float* __restrict__ xout, u16* __restrict__ xh) {
  int m = blockIdx.x;  // 0..M_-1
  const int tid = threadIdx.x;
  __shared__ float red[256];
  const float* yr = y + (ybcast ? (size_t)(m / S_) * E_ : (size_t)m * E_);
  const float* xr = xres + (size_t)m * E_;

  float v[4]; float sum = 0.f;
#pragma unroll
  for (int j = 0; j < 4; ++j) {
    int c = tid + j * 256;
    v[j] = xr[c] + yr[c]; sum += v[j];
  }
  red[tid] = sum; __syncthreads();
  for (int o = 128; o > 0; o >>= 1) { if (tid < o) red[tid] += red[tid + o]; __syncthreads(); }
  float mean = red[0] * (1.f / E_);
  __syncthreads();
  float sq = 0.f;
#pragma unroll
  for (int j = 0; j < 4; ++j) { float d = v[j] - mean; sq += d * d; }
  red[tid] = sq; __syncthreads();
  for (int o = 128; o > 0; o >>= 1) { if (tid < o) red[tid] += red[tid + o]; __syncthreads(); }
  float var = red[0] * (1.f / E_);
  float rstd = rsqrtf(var + 1e-5f);
#pragma unroll
  for (int j = 0; j < 4; ++j) {
    int c = tid + j * 256;
    float o = (v[j] - mean) * rstd * g[c] + bb[c];
    size_t ofs = (size_t)m * E_ + c;
    xout[ofs] = o; xh[ofs] = f2bf(o);
  }
}

// ---------------------------------------------------------------- launcher
extern "C" void kernel_launch(void* const* d_in, const int* in_sizes, int n_in,
                              void* d_out, int out_size, void* d_ws, size_t ws_size,
                              hipStream_t stream) {
  const float* features    = (const float*)d_in[0];
  const int*   captions    = (const int*)d_in[1];
  const float* embed_table = (const float*)d_in[2];
  const float* pos_enc     = (const float*)d_in[3];
  const float* sa_w  = (const float*)d_in[4];
  const float* sa_b  = (const float*)d_in[5];
  const float* sa_ow = (const float*)d_in[6];
  const float* sa_ob = (const float*)d_in[7];
  const float* ca_w  = (const float*)d_in[8];
  const float* ca_b  = (const float*)d_in[9];
  const float* ca_ow = (const float*)d_in[10];
  const float* ca_ob = (const float*)d_in[11];
  const float* ln_g  = (const float*)d_in[12];
  const float* ln_b  = (const float*)d_in[13];
  const float* ff_w1 = (const float*)d_in[14];
  const float* ff_b1 = (const float*)d_in[15];
  const float* ff_w2 = (const float*)d_in[16];
  const float* ff_b2 = (const float*)d_in[17];
  const float* out_w = (const float*)d_in[18];
  const float* out_b = (const float*)d_in[19];
  float* out = (float*)d_out;
  (void)in_sizes; (void)n_in; (void)out_size; (void)ws_size;

  char* wsp = (char*)d_ws; size_t off = 0;
  auto alloc = [&](size_t bytes) -> void* {
    void* p = wsp + off; off += (bytes + 255) & ~(size_t)255; return p;
  };
  float* xf   = (float*)alloc((size_t)MP_ * E_ * 4);
  u16*   xh   = (u16*)  alloc((size_t)MP_ * E_ * 2);
  float* qkv  = (float*)alloc((size_t)MP_ * 3 * E_ * 4);
  float* ctx  = (float*)alloc((size_t)MP_ * E_ * 4);
  u16*   ctxh = (u16*)  alloc((size_t)MP_ * E_ * 2);
  float* tmp  = (float*)alloc((size_t)MP_ * E_ * 4);
  u16*   ffhh = (u16*)  alloc((size_t)MP_ * FF_ * 2);  // FFN hidden, bf16 (fused relu)
  u16*   wb   = (u16*)  alloc((size_t)V_ * E_ * 2);    // bf16 weight staging (max: out_w)
  u16*   fh   = (u16*)  alloc((size_t)128 * E_ * 2);
  float* vm   = (float*)alloc((size_t)128 * E_ * 4);
  u16*   vmh  = (u16*)  alloc((size_t)128 * E_ * 2);
  float* cao  = (float*)alloc((size_t)128 * E_ * 4);

  auto cvt = [&](const float* src, u16* dst, long n) {
    unsigned g = (unsigned)((n + 1023) / 1024);
    cvt_f32_bf16<<<dim3(g), 256, 0, stream>>>(src, dst, n);
  };

  embed_kernel<<<MP_, 256, 0, stream>>>(captions, embed_table, pos_enc, xf, xh);
  feat_pad_kernel<<<128, 256, 0, stream>>>(features, fh);

  for (int l = 0; l < L_; ++l) {
    const float* sw  = sa_w  + (size_t)l * 3 * E_ * E_;
    const float* sb  = sa_b  + (size_t)l * 3 * E_;
    const float* sow = sa_ow + (size_t)l * E_ * E_;
    const float* sob = sa_ob + (size_t)l * E_;
    const float* cw  = ca_w  + (size_t)l * 3 * E_ * E_;
    const float* cb  = ca_b  + (size_t)l * 3 * E_;
    const float* cow = ca_ow + (size_t)l * E_ * E_;
    const float* cob = ca_ob + (size_t)l * E_;
    const float* g0  = ln_g + ((size_t)l * 3) * E_;
    const float* b0  = ln_b + ((size_t)l * 3) * E_;

    // ---- self attention
    cvt(sw, wb, (long)3 * E_ * E_);
    gemm_bf16_wmma<0><<<dim3(3 * E_ / 128, MP_ / 128), 256, 0, stream>>>(
        xh, wb, sb, qkv, nullptr, 3 * E_, E_, M_);
    attn_kernel<<<dim3(S_, NH_, B_), 128, 0, stream>>>(qkv, ctx, ctxh);
    cvt(sow, wb, (long)E_ * E_);
    gemm_bf16_wmma<0><<<dim3(E_ / 128, MP_ / 128), 256, 0, stream>>>(
        ctxh, wb, sob, tmp, nullptr, E_, E_, M_);
    add_ln_kernel<<<M_, 256, 0, stream>>>(xf, tmp, 0, g0, b0, xf, xh);

    // ---- cross attention: Sk==1 => softmax==1 => ctx = V(mem), broadcast over S
    cvt(cw + (size_t)2 * E_ * E_, wb, (long)E_ * E_);       // Wv slice of ca_w
    gemm_bf16_wmma<0><<<dim3(E_ / 128, 1), 256, 0, stream>>>(
        fh, wb, cb + 2 * E_, vm, nullptr, E_, E_, 128);
    cvt(vm, vmh, (long)128 * E_);
    cvt(cow, wb, (long)E_ * E_);
    gemm_bf16_wmma<0><<<dim3(E_ / 128, 1), 256, 0, stream>>>(
        vmh, wb, cob, cao, nullptr, E_, E_, 128);
    add_ln_kernel<<<M_, 256, 0, stream>>>(xf, cao, 1, g0 + E_, b0 + E_, xf, xh);

    // ---- FFN (ReLU + bf16 conversion fused into GEMM1 epilogue)
    cvt(ff_w1 + (size_t)l * FF_ * E_, wb, (long)FF_ * E_);
    gemm_bf16_wmma<1><<<dim3(FF_ / 128, MP_ / 128), 256, 0, stream>>>(
        xh, wb, ff_b1 + (size_t)l * FF_, nullptr, ffhh, FF_, E_, M_);
    cvt(ff_w2 + (size_t)l * E_ * FF_, wb, (long)E_ * FF_);
    gemm_bf16_wmma<0><<<dim3(E_ / 128, MP_ / 128), 256, 0, stream>>>(
        ffhh, wb, ff_b2 + (size_t)l * E_, tmp, nullptr, E_, FF_, M_);
    add_ln_kernel<<<M_, 256, 0, stream>>>(xf, tmp, 0, g0 + 2 * E_, b0 + 2 * E_, xf, xh);
  }

  // ---- vocab projection, scattered into out[:, 1:, :]; zero out[:, 0, :]
  cvt(out_w, wb, (long)V_ * E_);
  gemm_bf16_wmma<2><<<dim3(V_ / 128, MP_ / 128), 256, 0, stream>>>(
      xh, wb, out_b, out, nullptr, V_, E_, M_);
  zero_t0_kernel<<<(B_ * V_ + 255) / 256, 256, 0, stream>>>(out);
}
